// UniversalReasoner_16381005267442
// MI455X (gfx1250) — matrix operand order repaired and checked by reading, not compiled, not run on hardware
//
#include <hip/hip_runtime.h>
#include <hip/hip_bf16.h>
#include <math.h>

// ---------------------------------------------------------------------------
// UniversalReasoner for MI455X (gfx1250, wave32, WMMA + Tensor Data Mover).
// All GEMMs: bf16 inputs, f32 accumulate via v_wmma_f32_16x16x32_bf16.
// A tiles (and NT B tiles) staged into LDS by TDM (tensor_load_to_lds),
// overlapped with compute, synchronized with TENSORcnt + workgroup barrier.
// ---------------------------------------------------------------------------

typedef __attribute__((ext_vector_type(16))) __bf16 v16bf;
typedef __attribute__((ext_vector_type(8)))  float  v8f;
typedef __attribute__((ext_vector_type(4)))  unsigned int u32x4;
typedef __attribute__((ext_vector_type(8)))  int i32x8;
typedef __attribute__((ext_vector_type(4)))  int i32x4;

#define B_    4
#define S_    512
#define NS_   256
#define T_    768      // S_ + NS_
#define D_    768
#define H_    8
#define G_    2
#define HD_   96
#define DM_   3072     // 4*D
#define V_    50257
#define VP_   50304    // V_ padded to a multiple of 128 (aligned B tiles)
#define STEPS_ 4

// ---------------------------------------------------------------------------
// TDM: issue a 2D tile load (rows x d0 contiguous elements of 2 bytes,
// row stride in elements) from global memory into a contiguous LDS tile.
// Descriptor layout per CDNA5 ISA 8.3/8.4 (group0 128b, group1 256b, 2D).
// ---------------------------------------------------------------------------
__device__ __forceinline__ void tdm_load_2d(const void* gaddr, unsigned lds_addr,
                                            int tile_d0, int tile_d1,
                                            int tensor_d0, int tensor_d1,
                                            long long row_stride_elems)
{
  unsigned long long ga = (unsigned long long)gaddr;
  u32x4 g0;
  g0[0] = 1u;                                           // count=1, user-mode D#
  g0[1] = lds_addr;                                     // LDS byte address
  g0[2] = (unsigned)(ga & 0xFFFFFFFFu);                 // global_addr[31:0]
  g0[3] = (unsigned)((ga >> 32) & 0x01FFFFFFu)          // global_addr[56:32]
        | (2u << 30);                                   // type = 2 ("image")
  i32x8 g1;
  g1[0] = (1 << 16);                                    // data_size=1 (2B), no multicast
  g1[1] = (int)((tensor_d0 & 0xFFFF) << 16);            // tensor_dim0[15:0] @ b63:48
  g1[2] = (int)(((unsigned)tensor_d0 >> 16) & 0xFFFFu)  // tensor_dim0[31:16]
        | (int)((tensor_d1 & 0xFFFF) << 16);            // tensor_dim1[15:0]
  g1[3] = (int)(((unsigned)tensor_d1 >> 16) & 0xFFFFu)  // tensor_dim1[31:16]
        | (int)((tile_d0 & 0xFFFF) << 16);              // tile_dim0
  g1[4] = (int)(tile_d1 & 0xFFFF);                      // tile_dim1 (tile_dim2=0)
  g1[5] = (int)(row_stride_elems & 0xFFFFFFFFll);       // tensor_dim0_stride[31:0]
  g1[6] = (int)((row_stride_elems >> 32) & 0xFFFFll);   // stride[47:32]; dim1_stride=0
  g1[7] = 0;
  i32x4 zz = {0, 0, 0, 0};
#if defined(__clang_major__) && (__clang_major__ >= 23)
  i32x8 z8 = {0, 0, 0, 0, 0, 0, 0, 0};
  __builtin_amdgcn_tensor_load_to_lds(g0, g1, zz, zz, z8, 0);
#else
  __builtin_amdgcn_tensor_load_to_lds(g0, g1, zz, zz, 0);
#endif
}

__device__ __forceinline__ unsigned lds_off(const void* p) {
  return (unsigned)(unsigned long long)p;   // LDS aperture: addr[31:0] = LDS offset
}

// ---------------------------------------------------------------------------
// WMMA fragment loads from LDS tiles stored as [rows][32] bf16 (row = M or N,
// 32 = K slice). Layouts per CDNA5 ISA 7.12.2 (16-bit A 16x32; B striped).
// ---------------------------------------------------------------------------
__device__ __forceinline__ v16bf fragA16(const __bf16* basep, int rowBase, int lane) {
  const __bf16* p = basep + (size_t)(rowBase + (lane & 15)) * 32 + ((lane < 16) ? 0 : 8);
  union { v16bf v; uint4 q[2]; } u;
  u.q[0] = *(const uint4*)(p);
  u.q[1] = *(const uint4*)(p + 16);
  return u.v;
}

__device__ __forceinline__ v16bf fragB16(const __bf16* basep, int rowBase, int lane) {
  const __bf16* p = basep + (size_t)(rowBase + (lane & 15)) * 32 + ((lane < 16) ? 0 : 8);
  union { v16bf v; uint4 q[2]; } u;
  u.q[0] = *(const uint4*)(p);
  u.q[1] = *(const uint4*)(p + 16);
  return u.v;
}

__device__ __forceinline__ float gelu_tanh(float u) {
  return 0.5f * u * (1.0f + tanhf(0.7978845608f * (u + 0.044715f * u * u * u)));
}

// ---------------------------------------------------------------------------
// Generic batched WMMA GEMM.  C[m,n] = act(A.B + bias + Res)
//   BT=false : B is [K,N] row-major (transposed into LDS manually).
//   BT=true  : B is [N,K] row-major (NT gemm; TDM-loadable like A).
// Block tile 64x128, 8 waves of 32x32 (2x2 v_wmma_f32_16x16x32_bf16).
// Double-buffered LDS; A (and NT-B) tiles fetched by TDM from wave 0,
// overlapped with compute of the current tile.
// nb = number of loadable columns in B (>= N when padded) for the NN
// vectorized fast path.
// ---------------------------------------------------------------------------
template<bool BT>
__global__ __launch_bounds__(256)
void gemm_wmma(const __bf16* __restrict__ A, const __bf16* __restrict__ Bm,
               const float* __restrict__ bias, const float* __restrict__ Res,
               float* __restrict__ C, __bf16* __restrict__ Cb,
               int M, int N, int K, int lda, int ldb, int ldc, int nb,
               long long batA, long long batB, long long batC,
               int cdiv, long long batC2, int act)
{
  constexpr int BM = 64, BN = 128;
  __shared__ alignas(16) __bf16 As[2][BM][32];
  __shared__ alignas(16) __bf16 Bs[2][BN][32];

  const int z = blockIdx.z;
  A  += (long long)z * batA;
  Bm += (long long)z * batB;
  const long long coff = (long long)(z / cdiv) * batC + (long long)(z % cdiv) * batC2;
  if (C)   C   += coff;
  if (Cb)  Cb  += coff;
  if (Res) Res += coff;

  const int bm0 = blockIdx.y * BM;
  const int bn0 = blockIdx.x * BN;
  const int tid  = threadIdx.x;
  const int lane = tid & 31;
  const int wid  = tid >> 5;
  const int wm   = wid >> 2;   // 0..1
  const int wn   = wid & 3;    // 0..3

  v8f acc00 = {0.f,0.f,0.f,0.f,0.f,0.f,0.f,0.f};
  v8f acc01 = acc00, acc10 = acc00, acc11 = acc00;

  const int nk = K >> 5;       // K is always a multiple of 32 here
  int rowsA = M - bm0; if (rowsA > BM) rowsA = BM;
  int rowsB = N - bn0; if (rowsB > BN) rowsB = BN;

  // wave 0: issue async TDM tile loads for K-slice k0 into buffer 'buf'
  auto issueTDM = [&](int buf, int k0) {
    if (wid == 0) {
      tdm_load_2d(A + (long long)bm0 * lda + k0, lds_off(&As[buf][0][0]),
                  32, BM, 32, rowsA, (long long)lda);
      if constexpr (BT)
        tdm_load_2d(Bm + (long long)bn0 * ldb + k0, lds_off(&Bs[buf][0][0]),
                    32, BN, 32, rowsB, (long long)ldb);
    }
  };

  // all threads: manual transpose-fill of the NN B tile ([K,N] -> Bs[n][k])
  auto loadB_NN = [&](int buf, int k0) {
    if constexpr (!BT) {
      int kr  = tid >> 3;          // 0..31
      int nc0 = (tid & 7) * 16;    // 0..112
      const __bf16* prow = Bm + (long long)(k0 + kr) * ldb + bn0 + nc0;
      if (bn0 + BN <= nb) {        // uniform full-tile fast path (16B loads)
        union { uint4 q; __bf16 e[8]; } u0, u1;
        u0.q = *(const uint4*)(prow);
        u1.q = *(const uint4*)(prow + 8);
        __builtin_prefetch(prow + 32 * ldb, 0, 1);   // next K tile
        #pragma unroll
        for (int j = 0; j < 8; ++j) {
          Bs[buf][nc0 + j][kr]     = u0.e[j];
          Bs[buf][nc0 + 8 + j][kr] = u1.e[j];
        }
      } else {
        #pragma unroll
        for (int j = 0; j < 16; ++j) {
          int n = bn0 + nc0 + j;
          __bf16 v = (__bf16)0.0f;
          if (n < N) v = prow[j - 0 + 0 - nc0 - bn0 + n];  // == Bm[(k0+kr)*ldb + n + j]
          Bs[buf][nc0 + j][kr] = v;
        }
      }
    }
  };

  issueTDM(0, 0);
  loadB_NN(0, 0);
  if (wid == 0) __builtin_amdgcn_s_wait_tensorcnt(0);
  __syncthreads();

  for (int kt = 0; kt < nk; ++kt) {
    const int cur = kt & 1;
    if (kt + 1 < nk) {
      issueTDM(cur ^ 1, (kt + 1) << 5);
      loadB_NN(cur ^ 1, (kt + 1) << 5);
    }

    v16bf a0 = fragA16(&As[cur][0][0], wm * 32,      lane);
    v16bf a1 = fragA16(&As[cur][0][0], wm * 32 + 16, lane);
    v16bf b0 = fragB16(&Bs[cur][0][0], wn * 32,      lane);
    v16bf b1 = fragB16(&Bs[cur][0][0], wn * 32 + 16, lane);

    acc00 = __builtin_amdgcn_wmma_f32_16x16x32_bf16(false, a0, false, b0, (short)0, acc00, false, false);
    acc01 = __builtin_amdgcn_wmma_f32_16x16x32_bf16(false, a0, false, b1, (short)0, acc01, false, false);
    acc10 = __builtin_amdgcn_wmma_f32_16x16x32_bf16(false, a1, false, b0, (short)0, acc10, false, false);
    acc11 = __builtin_amdgcn_wmma_f32_16x16x32_bf16(false, a1, false, b1, (short)0, acc11, false, false);

    if (wid == 0) __builtin_amdgcn_s_wait_tensorcnt(0);
    __syncthreads();
  }

  // epilogue. C/D layout: VGPR r -> M = r + 8*(lane>=16), N = lane&15
  const int mlane = (lane >> 4) << 3;
  const int nlane = lane & 15;
  v8f accs[2][2] = { {acc00, acc01}, {acc10, acc11} };
  #pragma unroll
  for (int i = 0; i < 2; ++i) {
    #pragma unroll
    for (int j = 0; j < 2; ++j) {
      #pragma unroll
      for (int r = 0; r < 8; ++r) {
        int m = bm0 + wm * 32 + i * 16 + mlane + r;
        int n = bn0 + wn * 32 + j * 16 + nlane;
        if (m < M && n < N) {
          float v = accs[i][j][r];
          if (bias) v += bias[n];
          if (Res)  v += Res[(long long)m * ldc + n];
          if (act == 1) v = gelu_tanh(v);
          if (C)  C[(long long)m * ldc + n] = v;
          if (Cb) Cb[(long long)m * ldc + n] = (__bf16)v;
        }
      }
    }
  }
}

// ---------------------------------------------------------------------------
// Elementwise / reduction kernels
// ---------------------------------------------------------------------------
__global__ void k_f2bf(const float* __restrict__ in, __bf16* __restrict__ out, long long n) {
  long long i = (long long)blockIdx.x * blockDim.x + threadIdx.x;
  if (i < n) out[i] = (__bf16)in[i];
}

// convert with column padding: [rows,cols] f32 -> [rows,colsp] bf16 (zero pad)
__global__ void k_f2bf_pad(const float* __restrict__ in, __bf16* __restrict__ out,
                           int cols, int colsp, long long total) {
  long long i = (long long)blockIdx.x * blockDim.x + threadIdx.x;
  if (i >= total) return;
  int c = (int)(i % colsp);
  long long r = i / colsp;
  out[i] = (c < cols) ? (__bf16)in[r * cols + c] : (__bf16)0.0f;
}

__global__ void k_embed(const int* __restrict__ tokens, const float* __restrict__ embW,
                        const float* __restrict__ scratch, float* __restrict__ z) {
  long long i = (long long)blockIdx.x * blockDim.x + threadIdx.x;
  if (i >= (long long)B_ * T_ * D_) return;
  int d = (int)(i % D_);
  long long bt = i / D_;
  int t = (int)(bt % T_);
  int b = (int)(bt / T_);
  float v;
  if (t < S_) v = embW[(long long)tokens[b * S_ + t] * D_ + d];
  else        v = scratch[(long long)(t - S_) * D_ + d];
  z[i] = v;
}

__global__ __launch_bounds__(256)
void k_ln(const float* __restrict__ in, const float* __restrict__ gamma,
          const float* __restrict__ beta, __bf16* __restrict__ outb,
          const float* __restrict__ timeW, int step, float* __restrict__ xout)
{
  const int row = blockIdx.x;           // b*T + t
  const int t   = row % T_;
  const int tid = threadIdx.x;
  const float* r = in + (long long)row * D_;
  float loc[3]; float s = 0.f, s2 = 0.f;
  #pragma unroll
  for (int j = 0; j < 3; ++j) {
    int d = tid + j * 256;
    float v = r[d];
    if (timeW && t >= S_) v += timeW[step * D_ + d];
    loc[j] = v; s += v; s2 += v * v;
  }
  __shared__ float rs[256], rq[256];
  rs[tid] = s; rq[tid] = s2; __syncthreads();
  for (int off = 128; off > 0; off >>= 1) {
    if (tid < off) { rs[tid] += rs[tid + off]; rq[tid] += rq[tid + off]; }
    __syncthreads();
  }
  const float mu  = rs[0] * (1.0f / D_);
  const float var = rq[0] * (1.0f / D_) - mu * mu;
  const float inv = rsqrtf(var + 1e-6f);
  #pragma unroll
  for (int j = 0; j < 3; ++j) {
    int d = tid + j * 256;
    float v = loc[j];
    if (xout) xout[(long long)row * D_ + d] = v;
    outb[(long long)row * D_ + d] = (__bf16)((v - mu) * inv * gamma[d] + beta[d]);
  }
}

__global__ void k_rope_q(const float* __restrict__ qf, __bf16* __restrict__ qb) {
  const int d = threadIdx.x;            // 0..95
  int idx = blockIdx.x;                 // b*H*T
  int t  = idx % T_; int bh = idx / T_;
  int h  = bh % H_;  int b  = bh / H_;
  const float* src = qf + (((long long)(b * T_ + t)) * H_ + h) * HD_;
  float x  = src[d];
  float xr = (d < 48) ? -src[d + 48] : src[d - 48];
  float invf = __powf(10000.0f, -((float)(2 * (d % 48)) / (float)HD_));
  float ang = (float)t * invf;
  float c = cosf(ang), sn = sinf(ang);
  float scale = rsqrtf((float)HD_);
  qb[(((long long)(b * H_ + h)) * T_ + t) * HD_ + d] = (__bf16)((x * c + xr * sn) * scale);
}

__global__ void k_rope_kv(const float* __restrict__ kf, const float* __restrict__ vf,
                          __bf16* __restrict__ kb, __bf16* __restrict__ vb) {
  const int d = threadIdx.x;
  int idx = blockIdx.x;
  int t  = idx % T_; int bh = idx / T_;
  int h  = bh % H_;  int b  = bh / H_;
  int g  = h >> 2;                      // H/G = 4 heads per group
  const float* ks = kf + (((long long)(b * T_ + t)) * G_ + g) * HD_;
  const float* vs = vf + (((long long)(b * T_ + t)) * G_ + g) * HD_;
  float x  = ks[d];
  float xr = (d < 48) ? -ks[d + 48] : ks[d - 48];
  float invf = __powf(10000.0f, -((float)(2 * (d % 48)) / (float)HD_));
  float ang = (float)t * invf;
  float c = cosf(ang), sn = sinf(ang);
  long long o = (((long long)(b * H_ + h)) * T_ + t) * HD_ + d;
  kb[o] = (__bf16)(x * c + xr * sn);
  vb[o] = (__bf16)vs[d];
}

__global__ __launch_bounds__(256)
void k_softmax(const float* __restrict__ logits, __bf16* __restrict__ probs) {
  const int row = blockIdx.x;           // (b*H + h)*T + q
  const int q   = row % T_;
  const int tid = threadIdx.x;
  const float scale = rsqrtf((float)HD_);
  const float* L = logits + (long long)row * T_;
  float loc[3]; float mx = -3.0e38f;
  #pragma unroll
  for (int j = 0; j < 3; ++j) {
    int kc = tid + j * 256;
    bool ok = (q < S_) ? (kc <= q && kc < S_) : true;
    float v = ok ? (L[kc] * scale) : -1.0e30f;
    loc[j] = v; mx = fmaxf(mx, v);
  }
  __shared__ float red[256];
  red[tid] = mx; __syncthreads();
  for (int off = 128; off > 0; off >>= 1) { if (tid < off) red[tid] = fmaxf(red[tid], red[tid + off]); __syncthreads(); }
  mx = red[0]; __syncthreads();
  float s = 0.f;
  #pragma unroll
  for (int j = 0; j < 3; ++j) { loc[j] = __expf(loc[j] - mx); s += loc[j]; }
  red[tid] = s; __syncthreads();
  for (int off = 128; off > 0; off >>= 1) { if (tid < off) red[tid] += red[tid + off]; __syncthreads(); }
  const float invs = 1.0f / red[0];
  __bf16* P = probs + (long long)row * T_;
  #pragma unroll
  for (int j = 0; j < 3; ++j) P[tid + j * 256] = (__bf16)(loc[j] * invs);
}

__global__ __launch_bounds__(256)
void k_rowdot(const float* __restrict__ X, const float* __restrict__ w,
              const float* __restrict__ bias, int mode, float* __restrict__ out,
              int rowsPer, int rowOff)
{
  const int row = blockIdx.x;
  const int b = row / rowsPer, rl = row % rowsPer;
  const float* r = X + ((long long)b * T_ + rowOff + rl) * D_;
  const int tid = threadIdx.x;
  float s = 0.f;
  #pragma unroll
  for (int j = 0; j < 3; ++j) { int d = tid + j * 256; s += r[d] * w[d]; }
  __shared__ float red[256];
  red[tid] = s; __syncthreads();
  for (int off = 128; off > 0; off >>= 1) { if (tid < off) red[tid] += red[tid + off]; __syncthreads(); }
  if (tid == 0) {
    float v = red[0];
    if (mode == 0) { v += bias[0]; v = 1.0f / (1.0f + __expf(-v)); }
    out[row] = v;
  }
}

__global__ __launch_bounds__(256)
void k_update(float* __restrict__ zstate, const float* __restrict__ zraw,
              const float* __restrict__ sal, float* __restrict__ allz_step,
              float* __restrict__ tl_acc, float* __restrict__ sh_acc, int step)
{
  const int row = blockIdx.x;            // b*S + s
  const int b = row / S_, s2 = row % S_;
  const int tid = threadIdx.x;
  float*       cz = zstate + ((long long)b * T_ + s2) * D_;
  const float* nr = zraw   + ((long long)b * T_ + s2) * D_;
  float*       az = allz_step + (long long)row * D_;
  const float svl = sal[row];
  float tl = 0.f, sh = 0.f;
  #pragma unroll
  for (int j = 0; j < 3; ++j) {
    int d = tid + j * 256;
    float c = cz[d], n = nr[d];
    float df = n - c;
    float nv = c + svl * df;
    cz[d] = nv; az[d] = nv;
    tl += (1.0f - svl) * df * df;
    sh += fabsf(svl * df);
  }
  __shared__ float r1[256], r2[256];
  r1[tid] = tl; r2[tid] = sh; __syncthreads();
  for (int off = 128; off > 0; off >>= 1) {
    if (tid < off) { r1[tid] += r1[tid + off]; r2[tid] += r2[tid + off]; }
    __syncthreads();
  }
  if (tid == 0) {
    atomicAdd(&tl_acc[step * B_ + b], r1[0]);
    atomicAdd(&sh_acc[step * B_ + b], r2[0]);
  }
}

__global__ void k_copy_scratch(float* __restrict__ zstate, const float* __restrict__ zraw) {
  long long i = (long long)blockIdx.x * blockDim.x + threadIdx.x;
  if (i >= (long long)B_ * NS_ * D_) return;
  int d = (int)(i % D_);
  long long rr = i / D_;
  int ns = (int)(rr % NS_);
  int b  = (int)(rr / NS_);
  long long o = ((long long)b * T_ + S_ + ns) * D_ + d;
  zstate[o] = zraw[o];
}

__global__ __launch_bounds__(256)
void k_halt_reduce(const float* __restrict__ haltdot, const float* __restrict__ halt_b,
                   float* __restrict__ baseh, int step) {
  const int b = blockIdx.x, tid = threadIdx.x;
  __shared__ float red[256];
  red[tid] = haltdot[b * NS_ + tid];
  __syncthreads();
  for (int off = 128; off > 0; off >>= 1) { if (tid < off) red[tid] += red[tid + off]; __syncthreads(); }
  if (tid == 0) baseh[step * B_ + b] = red[0] / (float)NS_ + halt_b[0];
}

__global__ void k_finalize(const float* __restrict__ tl_acc, const float* __restrict__ sh_acc,
                           const float* __restrict__ baseh, float* __restrict__ stepw,
                           float* __restrict__ out_tail) {
  int b = threadIdx.x;
  if (b >= B_) return;
  float p = 1.0f, ponder = 0.f, temporal = 0.f;
  const float inv = 1.0f / ((float)S_ * (float)D_);
  for (int st = 0; st < STEPS_; ++st) {
    float shift = sh_acc[st * B_ + b] * inv;
    float halt  = 1.0f / (1.0f + __expf(-(baseh[st * B_ + b] - shift)));
    float tl    = tl_acc[st * B_ + b] * inv;
    float w = halt * p;
    if (st == STEPS_ - 1) w += p * (1.0f - halt);
    stepw[st * B_ + b] = w;
    ponder   += w * (float)(st + 1);
    temporal += w * tl;
    p *= (1.0f - halt);
  }
  out_tail[b]      = ponder;
  out_tail[B_ + b] = temporal;
}

__global__ void k_weighted(const float* __restrict__ allz, const float* __restrict__ stepw,
                           __bf16* __restrict__ wzb) {
  long long i = (long long)blockIdx.x * blockDim.x + threadIdx.x;
  if (i >= (long long)B_ * S_ * D_) return;
  int b = (int)(i / ((long long)S_ * D_));
  float acc = 0.f;
  #pragma unroll
  for (int st = 0; st < STEPS_; ++st)
    acc += stepw[st * B_ + b] * allz[(long long)st * B_ * S_ * D_ + i];
  wzb[i] = (__bf16)acc;
}

// ---------------------------------------------------------------------------
// Host orchestration
// ---------------------------------------------------------------------------
extern "C" void kernel_launch(void* const* d_in, const int* in_sizes, int n_in,
                              void* d_out, int out_size, void* d_ws, size_t ws_size,
                              hipStream_t stream)
{
  (void)in_sizes; (void)n_in; (void)out_size; (void)ws_size;

  const int*   tokens  = (const int*)  d_in[0];
  const float* embW    = (const float*)d_in[1];
  const float* timeW   = (const float*)d_in[2];
  const float* scratch = (const float*)d_in[3];
  const float* qW  = (const float*)d_in[4];  const float* qb_  = (const float*)d_in[5];
  const float* kW  = (const float*)d_in[6];  const float* kb_  = (const float*)d_in[7];
  const float* vW  = (const float*)d_in[8];  const float* vb_  = (const float*)d_in[9];
  const float* oW  = (const float*)d_in[10]; const float* ob_  = (const float*)d_in[11];
  const float* n1s = (const float*)d_in[12]; const float* n1b  = (const float*)d_in[13];
  const float* n2s = (const float*)d_in[14]; const float* n2b  = (const float*)d_in[15];
  const float* m1W = (const float*)d_in[16]; const float* m1b  = (const float*)d_in[17];
  const float* m2W = (const float*)d_in[18]; const float* m2b  = (const float*)d_in[19];
  const float* salW  = (const float*)d_in[20]; const float* salb  = (const float*)d_in[21];
  const float* haltW = (const float*)d_in[22]; const float* haltb = (const float*)d_in[23];
  const float* decW  = (const float*)d_in[24]; const float* decb  = (const float*)d_in[25];
  float* out = (float*)d_out;

  char* wsb = (char*)d_ws; size_t off = 0;
  auto alloc = [&](size_t bytes) -> void* {
    void* p = wsb + off;
    off += (bytes + 255) & ~(size_t)255;
    return p;
  };

  const long long BTD = (long long)B_ * T_ * D_;

  __bf16* qWb   = (__bf16*)alloc((size_t)D_ * D_  * 2);
  __bf16* kWb   = (__bf16*)alloc((size_t)D_ * 192 * 2);
  __bf16* vWb   = (__bf16*)alloc((size_t)D_ * 192 * 2);
  __bf16* oWb   = (__bf16*)alloc((size_t)D_ * D_  * 2);
  __bf16* m1Wb  = (__bf16*)alloc((size_t)D_ * DM_ * 2);
  __bf16* m2Wb  = (__bf16*)alloc((size_t)DM_ * D_ * 2);
  __bf16* decWb = (__bf16*)alloc((size_t)D_ * VP_ * 2);   // padded columns

  float*  zstate = (float*) alloc((size_t)BTD * 4);
  float*  xbuf   = (float*) alloc((size_t)BTD * 4);
  __bf16* hb     = (__bf16*)alloc((size_t)BTD * 2);
  float*  qf     = (float*) alloc((size_t)BTD * 4);   // q pre-rope; reused as x2
  float*  kf     = (float*) alloc((size_t)B_ * T_ * 192 * 4);
  float*  vf     = (float*) alloc((size_t)B_ * T_ * 192 * 4);
  __bf16* qbm    = (__bf16*)alloc((size_t)BTD * 2);   // [B,H,T,hd]
  __bf16* kbm    = (__bf16*)alloc((size_t)BTD * 2);
  __bf16* vbm    = (__bf16*)alloc((size_t)BTD * 2);
  float*  logitsf= (float*) alloc((size_t)B_ * H_ * T_ * T_ * 4);
  __bf16* probsb = (__bf16*)alloc((size_t)B_ * H_ * T_ * T_ * 2);
  __bf16* attnb  = (__bf16*)alloc((size_t)BTD * 2);
  __bf16* h2b    = (__bf16*)alloc((size_t)BTD * 2);
  __bf16* gb     = (__bf16*)alloc((size_t)B_ * T_ * DM_ * 2);
  float*  zraw   = (float*) alloc((size_t)BTD * 4);
  float*  salv   = (float*) alloc((size_t)B_ * S_ * 4);
  float*  haltd  = (float*) alloc((size_t)B_ * NS_ * 4);
  float*  accs   = (float*) alloc((size_t)3 * STEPS_ * B_ * 4);
  float*  tl_acc = accs;
  float*  sh_acc = accs + STEPS_ * B_;
  float*  baseh  = accs + 2 * STEPS_ * B_;
  float*  stepw  = (float*) alloc((size_t)STEPS_ * B_ * 4);
  float*  allz   = (float*) alloc((size_t)STEPS_ * B_ * S_ * D_ * 4);
  __bf16* wzb    = (__bf16*)alloc((size_t)B_ * S_ * D_ * 2);

  float* x2 = qf;

  auto conv = [&](const float* src, __bf16* dst, long long n) {
    k_f2bf<<<dim3((unsigned)((n + 255) / 256)), 256, 0, stream>>>(src, dst, n);
  };
  conv(qW,  qWb,  (long long)D_ * D_);
  conv(kW,  kWb,  (long long)D_ * 192);
  conv(vW,  vWb,  (long long)D_ * 192);
  conv(oW,  oWb,  (long long)D_ * D_);
  conv(m1W, m1Wb, (long long)D_ * DM_);
  conv(m2W, m2Wb, (long long)DM_ * D_);
  {
    long long tot = (long long)D_ * VP_;
    k_f2bf_pad<<<dim3((unsigned)((tot + 255) / 256)), 256, 0, stream>>>(decW, decWb, V_, VP_, tot);
  }

  k_embed<<<dim3((unsigned)((BTD + 255) / 256)), 256, 0, stream>>>(tokens, embW, scratch, zstate);

  hipMemsetAsync(accs, 0, (size_t)3 * STEPS_ * B_ * sizeof(float), stream);

  const long long sTHD = (long long)T_ * HD_;
  const long long sTT  = (long long)T_ * T_;
  const long long sTD  = (long long)T_ * D_;

  for (int step = 0; step < STEPS_; ++step) {
    k_ln<<<B_ * T_, 256, 0, stream>>>(zstate, n1s, n1b, hb, timeW, step, xbuf);

    gemm_wmma<false><<<dim3(6, 48, 1), 256, 0, stream>>>(hb, qWb, qb_, nullptr, qf, nullptr,
        B_ * T_, D_, D_, D_, D_, D_, D_, 0, 0, 0, 1, 0, 0);
    gemm_wmma<false><<<dim3(2, 48, 1), 256, 0, stream>>>(hb, kWb, kb_, nullptr, kf, nullptr,
        B_ * T_, 192, D_, D_, 192, 192, 192, 0, 0, 0, 1, 0, 0);
    gemm_wmma<false><<<dim3(2, 48, 1), 256, 0, stream>>>(hb, vWb, vb_, nullptr, vf, nullptr,
        B_ * T_, 192, D_, D_, 192, 192, 192, 0, 0, 0, 1, 0, 0);

    k_rope_q <<<B_ * H_ * T_, 96, 0, stream>>>(qf, qbm);
    k_rope_kv<<<B_ * H_ * T_, 96, 0, stream>>>(kf, vf, kbm, vbm);

    // logits = Q x K^T (batched per b,h), NT gemm: A and B tiles both via TDM
    gemm_wmma<true><<<dim3(6, 12, 32), 256, 0, stream>>>(qbm, kbm, nullptr, nullptr, logitsf, nullptr,
        T_, T_, HD_, HD_, HD_, T_, T_, sTHD, sTHD, sTT, 1, 0, 0);

    k_softmax<<<B_ * H_ * T_, 256, 0, stream>>>(logitsf, probsb);

    gemm_wmma<false><<<dim3(1, 12, 32), 256, 0, stream>>>(probsb, vbm, nullptr, nullptr, nullptr, attnb,
        T_, HD_, T_, T_, HD_, D_, HD_, sTT, sTHD, sTD, H_, HD_, 0);

    gemm_wmma<false><<<dim3(6, 48, 1), 256, 0, stream>>>(attnb, oWb, ob_, xbuf, x2, nullptr,
        B_ * T_, D_, D_, D_, D_, D_, D_, 0, 0, 0, 1, 0, 0);

    k_ln<<<B_ * T_, 256, 0, stream>>>(x2, n2s, n2b, h2b, nullptr, 0, nullptr);

    gemm_wmma<false><<<dim3(24, 48, 1), 256, 0, stream>>>(h2b, m1Wb, m1b, nullptr, nullptr, gb,
        B_ * T_, DM_, D_, D_, DM_, DM_, DM_, 0, 0, 0, 1, 0, 1);

    gemm_wmma<false><<<dim3(6, 48, 1), 256, 0, stream>>>(gb, m2Wb, m2b, x2, zraw, nullptr,
        B_ * T_, D_, DM_, DM_, D_, D_, D_, 0, 0, 0, 1, 0, 0);

    k_rowdot<<<B_ * S_,  256, 0, stream>>>(zstate, salW,  salb,  0, salv,  S_,  0);
    k_rowdot<<<B_ * NS_, 256, 0, stream>>>(zraw,   haltW, haltb, 1, haltd, NS_, S_);

    k_update<<<B_ * S_, 256, 0, stream>>>(zstate, zraw, salv,
        allz + (long long)step * B_ * S_ * D_, tl_acc, sh_acc, step);
    k_copy_scratch<<<dim3((unsigned)(((long long)B_ * NS_ * D_ + 255) / 256)), 256, 0, stream>>>(zstate, zraw);
    k_halt_reduce<<<B_, 256, 0, stream>>>(haltd, haltb, baseh, step);
  }

  k_finalize<<<1, 64, 0, stream>>>(tl_acc, sh_acc, baseh, stepw, out + (long long)B_ * S_ * V_);

  k_weighted<<<dim3((unsigned)(((long long)B_ * S_ * D_ + 255) / 256)), 256, 0, stream>>>(allz, stepw, wzb);

  // decode: [2048,768] x [768,50257(pad 50304)] + dec_b -> d_out (f32)
  gemm_wmma<false><<<dim3(VP_ / 128, 32, 1), 256, 0, stream>>>(wzb, decWb, decb, nullptr, out, nullptr,
      B_ * S_, V_, D_, D_, VP_, V_, VP_, 0, 0, 0, 1, 0, 0);
}
